// Multi_Heads_Self_Attn_83021717832739
// MI455X (gfx1250) — matrix operand order repaired and checked by reading, not compiled, not run on hardware
//
#include <hip/hip_runtime.h>
#include <hip/hip_fp16.h>
#include <stdint.h>

// ---------- types ----------
typedef __attribute__((ext_vector_type(16))) _Float16 v16h;  // 8 VGPRs: WMMA A/B frag
typedef __attribute__((ext_vector_type(8)))  float    v8f;   // 8 VGPRs: WMMA C/D frag
typedef __attribute__((ext_vector_type(4)))  unsigned int uvec4; // POD 16B vector

union Frag16 { v16h h; uvec4 q[2]; uint32_t u[8]; _Float16 e[16]; };

#define NVOX 4096   // 16*16*16
#define CCH  128
#define HEADS 4
#define DKD  16
#define HD   64

// =====================================================================
// Kernel 1: QKV projection.  y[o,n] = sum_c W[o,c] * x[b,c,n]
// Q pre-scaled by dk^-0.5 = 0.25.  Output f16, layout [b][h][n][16].
// =====================================================================
__global__ void __launch_bounds__(256) qkv_kernel(
    const float* __restrict__ x,
    const float* __restrict__ Wq, const float* __restrict__ Wk,
    const float* __restrict__ Wv,
    _Float16* __restrict__ Qb, _Float16* __restrict__ Kb,
    _Float16* __restrict__ Vb)
{
    __shared__ float xs[CCH * 64];           // 32 KB: x tile [128 c][64 n]
    const int b  = blockIdx.y;
    const int n0 = blockIdx.x * 64;
    const int tid = threadIdx.x;
    const float* xb = x + (size_t)b * CCH * NVOX;

    #pragma unroll
    for (int i = 0; i < 32; ++i) {
        int idx = i * 256 + tid;
        int c = idx >> 6, n = idx & 63;
        xs[idx] = xb[(size_t)c * NVOX + n0 + n];   // coalesced: 64 consec per c-row
    }
    __syncthreads();

    const int n = tid & 63;       // voxel within tile
    const int g = tid >> 6;       // head (0..3), uniform per wave -> scalar W loads

    float acc[48];
    #pragma unroll
    for (int k = 0; k < 48; ++k) acc[k] = 0.f;

    const float* Ws[3] = { Wq, Wk, Wv };
    for (int cb = 0; cb < CCH; cb += 16) {
        float xr[16];
        #pragma unroll
        for (int cc = 0; cc < 16; ++cc) xr[cc] = xs[(cb + cc) * 64 + n];
        #pragma unroll
        for (int mm = 0; mm < 3; ++mm) {
            const float* Wm = Ws[mm];
            #pragma unroll
            for (int oo = 0; oo < 16; ++oo) {
                const float* wr = Wm + (size_t)(g * 16 + oo) * CCH + cb;
                float a = acc[mm * 16 + oo];
                #pragma unroll
                for (int cc = 0; cc < 16; ++cc) a += wr[cc] * xr[cc];
                acc[mm * 16 + oo] = a;
            }
        }
    }

    const size_t base = ((size_t)(b * HEADS + g) * NVOX + n0 + n) * DKD;
    #pragma unroll
    for (int oo = 0; oo < 16; ++oo) {
        Qb[base + oo] = (_Float16)(acc[oo] * 0.25f);       // dk^-0.5
        Kb[base + oo] = (_Float16)acc[16 + oo];
        Vb[base + oo] = (_Float16)acc[32 + oo];
    }
}

// =====================================================================
// Kernel 2: transposed flash attention.  One block = 4 waves = 64 queries.
// Per wave: 16 queries.  Per 32-key step:
//   S^T = K_tile x Q^T   (2x v_wmma; A = K rows from global, B = Q^T, d padded)
//   online softmax: per-lane (fixed query) VALU tree + ONE xor-16 shuffle each
//   P^T B-frag assembled in-register via 8 packed xor-16 half-exchanges
//   O^T += V^T x P^T     (1x v_wmma; A = V^T from LDS-transposed tile)
// =====================================================================
__global__ void __launch_bounds__(128) flash_kernel(
    const _Float16* __restrict__ Qb, const _Float16* __restrict__ Kb,
    const _Float16* __restrict__ Vb, float* __restrict__ merged)
{
    const int bh   = blockIdx.y;                 // b*4 + h
    const int lane = threadIdx.x & 31;
    const int wave = threadIdx.x >> 5;
    const int half = lane >> 4;                  // 0 / 1
    const int qcol = lane & 15;                  // this lane's query (C-layout col)

    const _Float16* Qh = Qb + (size_t)bh * NVOX * DKD;
    const _Float16* Kh = Kb + (size_t)bh * NVOX * DKD;
    const _Float16* Vh = Vb + (size_t)bh * NVOX * DKD;

    __shared__ __align__(16) _Float16 VT[16 * 32];     // V tile transposed [dk][key]

    const uvec4 z4 = { 0u, 0u, 0u, 0u };

    // ---- B-frag: Q^T [32d x 16q].  lane col = query; elems 8-15 (padded d16-31) = 0 in both halves.
    Frag16 qfb;
    {
        const int qrow = blockIdx.x * 64 + wave * 16 + qcol;
        {
            const uvec4* qp = (const uvec4*)(Qh + (size_t)qrow * DKD);
            qfb.q[0] = qp[half];       // d 0-7 / 8-15
            qfb.q[1] = z4;             // padded d 16-31
        }
    }

    v8f O = {};
    float m = -3.0e38f, l = 0.f;

    for (int kt = 0; kt < NVOX; kt += 32) {
        __syncthreads();                       // all waves done reading previous VT
        {   // cooperative transpose-fill of V tile: 256 u32 / 128 threads
            const uint32_t* vs = (const uint32_t*)(Vh + (size_t)kt * DKD);
            #pragma unroll
            for (int i = 0; i < 2; ++i) {
                int e = threadIdx.x + i * 128;         // u32 index 0..255
                union { uint32_t u; _Float16 h2[2]; } cv; cv.u = vs[e];
                int key = e >> 3, c0 = (e & 7) * 2;    // u32 = dk pair of one key
                VT[c0 * 32 + key]       = cv.h2[0];
                VT[(c0 + 1) * 32 + key] = cv.h2[1];
            }
        }
        __syncthreads();

        // ---- A-frags: K rows (16-bit A 16x32 layout, d 16-31 zero-padded)
        Frag16 ka0, ka1;
        {
            const _Float16* kp = Kh + (size_t)(kt + qcol) * DKD + half * 8;
            ka0.q[0] = *(const uvec4*)kp;                 // keys kt+0..15
            ka0.q[1] = z4;
            ka1.q[0] = *(const uvec4*)(kp + 16 * DKD);    // keys kt+16..31
            ka1.q[1] = z4;
            if (kt + 32 < NVOX) {                          // prefetch next tiles
                __builtin_prefetch(kp + 32 * DKD, 0, 1);
                __builtin_prefetch(Vh + (size_t)(kt + 32) * DKD + threadIdx.x * 4, 0, 1);
            }
        }
        v8f zc = {};
        v8f S0 = __builtin_amdgcn_wmma_f32_16x16x32_f16(false, ka0.h, false, qfb.h,
                                                        (short)0, zc, false, false);
        v8f S1 = __builtin_amdgcn_wmma_f32_16x16x32_f16(false, ka1.h, false, qfb.h,
                                                        (short)0, zc, false, false);
        // lane holds S^T[key][q]: keys j+8*half (tile0) and 16+j+8*half (tile1), q=qcol

        // ---- online softmax: per-lane tree + one xor-16 combine
        float cm = fmaxf(S0[0], S1[0]);
        #pragma unroll
        for (int j = 1; j < 8; ++j) cm = fmaxf(cm, fmaxf(S0[j], S1[j]));
        cm = fmaxf(cm, __shfl_xor(cm, 16, 32));
        const float mn = fmaxf(m, cm);
        const float sc = __expf(m - mn);

        float p0[8], p1[8], rs = 0.f;
        #pragma unroll
        for (int j = 0; j < 8; ++j) {
            p0[j] = __expf(S0[j] - mn);
            p1[j] = __expf(S1[j] - mn);
            rs += p0[j] + p1[j];
        }
        rs += __shfl_xor(rs, 16, 32);
        l = l * sc + rs;
        m = mn;
        #pragma unroll
        for (int j = 0; j < 8; ++j) O[j] *= sc;

        // ---- assemble P^T B-frag [32k x 16q] in-register:
        // lane<16 needs keys 0..15 (own p0 = k0-7, partner's p0 = k8-15)
        // lane>=16 needs keys 16..31 (partner's p1 = k16-23, own p1 = k24-31)
        Frag16 pfb;
        {
            uint32_t pA[4], pB[4];
            #pragma unroll
            for (int j = 0; j < 4; ++j) {
                union { uint32_t u; _Float16 h2[2]; } a, b;
                a.h2[0] = (_Float16)p0[2 * j]; a.h2[1] = (_Float16)p0[2 * j + 1];
                b.h2[0] = (_Float16)p1[2 * j]; b.h2[1] = (_Float16)p1[2 * j + 1];
                pA[j] = a.u; pB[j] = b.u;
            }
            #pragma unroll
            for (int j = 0; j < 4; ++j) {
                uint32_t rA = (uint32_t)__shfl_xor((int)pA[j], 16, 32);
                uint32_t rB = (uint32_t)__shfl_xor((int)pB[j], 16, 32);
                if (lane < 16) { pfb.u[j] = pA[j]; pfb.u[4 + j] = rA; }
                else           { pfb.u[j] = rB;    pfb.u[4 + j] = pB[j]; }
            }
        }

        // ---- A-frag: V^T rows (dk) from LDS-transposed tile
        Frag16 va;
        {
            const int seg = half * 16;
            va.q[0] = *(const uvec4*)&VT[qcol * 32 + seg];        // keys 0-7 / 16-23
            va.q[1] = *(const uvec4*)&VT[qcol * 32 + 8 + seg];    // keys 8-15 / 24-31
        }
        O = __builtin_amdgcn_wmma_f32_16x16x32_f16(false, va.h, false, pfb.h,
                                                   (short)0, O, false, false);
        // O^T layout: row dk = j+8*half, col q = qcol  -> scale/l are per-lane scalars
    }

    // ---- epilogue: O^T / l -> merged[b][n][h*16+dk], 8 contiguous floats per lane
    const float inv = 1.0f / l;
    const int b = bh >> 2, h = bh & 3;
    const int nrow = blockIdx.x * 64 + wave * 16 + qcol;
    float* dst = merged + ((size_t)(b * NVOX + nrow)) * HD + h * 16 + half * 8;
    float4 lo = make_float4(O[0] * inv, O[1] * inv, O[2] * inv, O[3] * inv);
    float4 hi = make_float4(O[4] * inv, O[5] * inv, O[6] * inv, O[7] * inv);
    *(float4*)dst       = lo;
    *(float4*)(dst + 4) = hi;
}

// =====================================================================
// Kernel 3: output projection + InstanceNorm(affine=False) + residual + ReLU
// one block per (b,c); mean/var over N=4096 via block reduction.
// =====================================================================
__global__ void __launch_bounds__(256) proj_norm_kernel(
    const float* __restrict__ merged, const float* __restrict__ Wo,
    const float* __restrict__ x, float* __restrict__ out)
{
    const int b = blockIdx.x >> 7, c = blockIdx.x & 127, tid = threadIdx.x;
    __shared__ float WoS[HD];
    __shared__ float red[512];
    if (tid < HD) WoS[tid] = Wo[(size_t)c * HD + tid];
    __syncthreads();

    float vals[16];
    float s_sum = 0.f, s_sq = 0.f;
    #pragma unroll
    for (int i = 0; i < 16; ++i) {
        int n = i * 256 + tid;
        const float4* mr = (const float4*)(merged + ((size_t)(b * NVOX + n)) * HD);
        float acc = 0.f;
        #pragma unroll
        for (int o4 = 0; o4 < 16; ++o4) {
            float4 mv = mr[o4];
            acc += WoS[o4 * 4 + 0] * mv.x + WoS[o4 * 4 + 1] * mv.y
                 + WoS[o4 * 4 + 2] * mv.z + WoS[o4 * 4 + 3] * mv.w;
        }
        vals[i] = acc; s_sum += acc; s_sq += acc * acc;
    }
    red[tid] = s_sum; red[256 + tid] = s_sq;
    __syncthreads();
    for (int st = 128; st > 0; st >>= 1) {
        if (tid < st) { red[tid] += red[tid + st]; red[256 + tid] += red[256 + tid + st]; }
        __syncthreads();
    }
    const float mean = red[0] * (1.0f / NVOX);
    const float var  = red[256] * (1.0f / NVOX) - mean * mean;
    const float rinv = rsqrtf(var + 1e-5f);

    const float* xr  = x   + (size_t)blockIdx.x * NVOX;   // (b*128+c)*4096
    float*       orr = out + (size_t)blockIdx.x * NVOX;
    #pragma unroll
    for (int i = 0; i < 16; ++i) {
        int n = i * 256 + tid;
        float y = (vals[i] - mean) * rinv + xr[n];
        orr[n] = fmaxf(y, 0.0f);
    }
}

// =====================================================================
extern "C" void kernel_launch(void* const* d_in, const int* in_sizes, int n_in,
                              void* d_out, int out_size, void* d_ws, size_t ws_size,
                              hipStream_t stream)
{
    (void)in_sizes; (void)n_in; (void)out_size; (void)ws_size;
    const float* x  = (const float*)d_in[0];
    const float* Wq = (const float*)d_in[1];
    const float* Wk = (const float*)d_in[2];
    const float* Wv = (const float*)d_in[3];
    const float* Wo = (const float*)d_in[4];
    float* out = (float*)d_out;

    const size_t QKV = (size_t)2 * HEADS * NVOX * DKD;   // 524288 halves each
    _Float16* Qb = (_Float16*)d_ws;
    _Float16* Kb = Qb + QKV;
    _Float16* Vb = Kb + QKV;
    float* merged = (float*)(Vb + QKV);                  // 2*4096*64 f32

    qkv_kernel<<<dim3(NVOX / 64, 2), 256, 0, stream>>>(x, Wq, Wk, Wv, Qb, Kb, Vb);
    flash_kernel<<<dim3(NVOX / 64, 2 * HEADS), 128, 0, stream>>>(Qb, Kb, Vb, merged);
    proj_norm_kernel<<<dim3(2 * CCH), 256, 0, stream>>>(merged, Wo, x, out);
}
